// MultiLayerPatchCoreModel_9792525435029
// MI455X (gfx1250) — matrix operand order repaired and checked by reading, not compiled, not run on hardware
//
#include <hip/hip_runtime.h>

typedef __bf16 bf16_t;
typedef __attribute__((ext_vector_type(16))) __bf16 v16bf;
typedef __attribute__((ext_vector_type(8)))  __bf16 v8bf;
typedef __attribute__((ext_vector_type(8)))  float  v8f;

// exact pointee type the async-LDS builtin wants: int __vector(4)
typedef int v4i __attribute__((__vector_size__(16)));
typedef v4i __attribute__((address_space(1))) gv4i;   // global
typedef v4i __attribute__((address_space(3))) lv4i;   // LDS

#define DIM       1536
#define NQ        6272     // 32 * 196
#define NM        65536
#define NB        32
#define NP        196
#define KTOP      20       // ceil(196 * 0.1)
#define COLS_PER_BLOCK 1024
#define MCHUNKS   (NM / COLS_PER_BLOCK)  // 64
#define QTILES    (NQ / 16)              // 392
#define NTILES_PER_WAVE 4               // each wave: 16 x 64 output tile

// ---------------------------------------------------------------------------
// L2-normalize rows of src (fp32) -> dst (bf16).  One block per row, 8 waves.
// ---------------------------------------------------------------------------
__global__ __launch_bounds__(256) void normalize_rows(const float* __restrict__ src,
                                                      bf16_t* __restrict__ dst,
                                                      int rows) {
    int row = blockIdx.x;
    if (row >= rows) return;
    const float* s = src + (size_t)row * DIM;
    bf16_t*      d = dst + (size_t)row * DIM;

    float ss = 0.0f;
    for (int i = threadIdx.x; i < DIM; i += blockDim.x) {
        float v = s[i];
        ss += v * v;
    }
    #pragma unroll
    for (int off = 16; off > 0; off >>= 1) ss += __shfl_xor(ss, off, 32);

    __shared__ float wsum[8];
    int wave = threadIdx.x >> 5;
    int lane = threadIdx.x & 31;
    if (lane == 0) wsum[wave] = ss;
    __syncthreads();

    float total = 0.0f;
    #pragma unroll
    for (int i = 0; i < 8; ++i) total += wsum[i];

    float scale = 1.0f / fmaxf(sqrtf(total), 1e-12f);
    for (int i = threadIdx.x; i < DIM; i += blockDim.x)
        d[i] = (bf16_t)(s[i] * scale);
}

// ---------------------------------------------------------------------------
// Init per-query min(d^2) to +inf (as monotonic uint bits).
// ---------------------------------------------------------------------------
__global__ void init_min(unsigned* __restrict__ qmin, int n) {
    int i = blockIdx.x * blockDim.x + threadIdx.x;
    if (i < n) qmin[i] = 0x7F800000u;  // +inf
}

// ---------------------------------------------------------------------------
// Core: 16-query x 1024-bank-column tile per block; 8 waves each own a
// 16 x 64 sub-tile (4 accumulators) so one A fragment feeds 4 WMMAs.
// A tile staged HBM->LDS with ASYNCcnt-tracked async copies; B streamed
// global->VGPR (row-major layout == B-fragment lane layout).
// ---------------------------------------------------------------------------
__global__ __launch_bounds__(256) void patchcore_nn(const bf16_t* __restrict__ qn,
                                                    const bf16_t* __restrict__ bn,
                                                    unsigned* __restrict__ qmin) {
    __shared__ bf16_t Atile[16 * DIM];  // 48 KB of the 320 KB WGP LDS budget

    const int qbase = blockIdx.x * 16;
    const int cbase = blockIdx.y * COLS_PER_BLOCK;

    // ---- Stage the 16x1536 bf16 query tile into LDS (async HBM->LDS) ----
    {
        const char* gA = (const char*)(qn + (size_t)qbase * DIM);
        char*       lA = (char*)Atile;
        const int nvec = 16 * DIM * 2 / 16;  // 3072 x 16B
        for (int i = threadIdx.x; i < nvec; i += blockDim.x) {
#if __has_builtin(__builtin_amdgcn_global_load_async_to_lds_b128)
            __builtin_amdgcn_global_load_async_to_lds_b128(
                (gv4i*)(uintptr_t)(gA + (size_t)i * 16),
                (lv4i*)(unsigned)(uintptr_t)(lA + (size_t)i * 16),
                /*offset=*/0, /*cpol=*/0);
#else
            ((uint4*)lA)[i] = ((const uint4*)gA)[i];
#endif
        }
#if __has_builtin(__builtin_amdgcn_s_wait_asynccnt)
        __builtin_amdgcn_s_wait_asynccnt(0);
#elif __has_builtin(__builtin_amdgcn_global_load_async_to_lds_b128)
        asm volatile("s_wait_asynccnt 0x0" ::: "memory");
#endif
    }
    __syncthreads();

    const int wave  = threadIdx.x >> 5;
    const int lane  = threadIdx.x & 31;
    const int lhalf = lane >> 4;   // 0: lanes 0-15, 1: lanes 16-31
    const int l16   = lane & 15;

    // A fragment source row in LDS (k order this code uses for A and B: lanes 0-15 hold
    // K {0..15}, lanes 16-31 hold K {16..31}).
    const bf16_t* arow  = Atile + l16 * DIM;
    const int     a_off = 16 * lhalf;

    float dmin[8];
    #pragma unroll
    for (int r = 0; r < 8; ++r) dmin[r] = 3.0e38f;

    for (int cg = 0; cg < COLS_PER_BLOCK; cg += 128 * NTILES_PER_WAVE) {
        // Lane's B column for sub-tile t: N = l16 within each 16-col tile.
        const bf16_t* brow[NTILES_PER_WAVE];
        #pragma unroll
        for (int t = 0; t < NTILES_PER_WAVE; ++t) {
            int col = cbase + cg + wave * (16 * NTILES_PER_WAVE) + t * 16 + l16;
            brow[t] = bn + (size_t)col * DIM + 16 * lhalf;
            // speculative prefetch of the next column group's stream
            if (cg + 128 * NTILES_PER_WAVE < COLS_PER_BLOCK)
                __builtin_prefetch(brow[t] + (size_t)128 * NTILES_PER_WAVE * DIM, 0, 1);
        }

        v8f acc[NTILES_PER_WAVE];
        #pragma unroll
        for (int t = 0; t < NTILES_PER_WAVE; ++t) acc[t] = (v8f){};

        #pragma unroll 2
        for (int k = 0; k < DIM; k += 32) {
            v8bf a0 = *(const v8bf*)(arow + k + a_off);
            v8bf a1 = *(const v8bf*)(arow + k + 8 + a_off);
            v16bf a;
            #pragma unroll
            for (int i = 0; i < 8; ++i) { a[i] = a0[i]; a[8 + i] = a1[i]; }

            #pragma unroll
            for (int t = 0; t < NTILES_PER_WAVE; ++t) {
                v16bf b = *(const v16bf*)(brow[t] + k);
                acc[t] = __builtin_amdgcn_wmma_f32_16x16x32_bf16(
                    /*neg_a=*/false, a, /*neg_b=*/false, b,
                    /*c_mod=*/(short)0, acc[t], /*reuse_a=*/false, /*reuse_b=*/false);
            }
        }

        #pragma unroll
        for (int t = 0; t < NTILES_PER_WAVE; ++t)
            #pragma unroll
            for (int r = 0; r < 8; ++r) {
                float d2 = fmaxf(2.0f - 2.0f * acc[t][r], 0.0f);
                dmin[r] = fminf(dmin[r], d2);
            }
    }

    // Min over the 16 columns held by each 16-lane group (xor masks < 16 keep
    // the two half-wave row groups separate).
    #pragma unroll
    for (int r = 0; r < 8; ++r) {
        float v = dmin[r];
        v = fminf(v, __shfl_xor(v, 1, 32));
        v = fminf(v, __shfl_xor(v, 2, 32));
        v = fminf(v, __shfl_xor(v, 4, 32));
        v = fminf(v, __shfl_xor(v, 8, 32));
        dmin[r] = v;
    }

    // C/D layout: VGPR r -> row r (lanes 0-15) / row r+8 (lanes 16-31).
    if (l16 == 0) {
        #pragma unroll
        for (int r = 0; r < 8; ++r) {
            int row = qbase + r + 8 * lhalf;
            atomicMin(qmin + row, __float_as_uint(dmin[r]));  // d2 >= 0: bits monotone
        }
    }
}

// ---------------------------------------------------------------------------
// Per-image top-20 mean of the 196 nearest distances.
// ---------------------------------------------------------------------------
__global__ __launch_bounds__(256) void topk_mean(const unsigned* __restrict__ qmin,
                                                 float* __restrict__ out) {
    __shared__ float dist[NP];
    const int b = blockIdx.x;
    for (int i = threadIdx.x; i < NP; i += blockDim.x)
        dist[i] = sqrtf(__uint_as_float(qmin[b * NP + i]));
    __syncthreads();

    if (threadIdx.x == 0) {
        float sum = 0.0f;
        for (int t = 0; t < KTOP; ++t) {
            int bi = 0; float bv = -1.0f;
            for (int i = 0; i < NP; ++i)
                if (dist[i] > bv) { bv = dist[i]; bi = i; }
            sum += bv;
            dist[bi] = -2.0f;
        }
        out[b] = sum / (float)KTOP;
    }
}

// ---------------------------------------------------------------------------
extern "C" void kernel_launch(void* const* d_in, const int* in_sizes, int n_in,
                              void* d_out, int out_size, void* d_ws, size_t ws_size,
                              hipStream_t stream) {
    const float* patch = (const float*)d_in[0];   // [32,196,1536] fp32
    const float* bank  = (const float*)d_in[1];   // [65536,1536]  fp32
    float* out = (float*)d_out;                   // [32] fp32

    char* ws = (char*)d_ws;
    bf16_t* bn = (bf16_t*)ws;                                          // 201.3 MB
    bf16_t* qn = (bf16_t*)(ws + (size_t)NM * DIM * 2);                 // 19.3 MB
    unsigned* qmin = (unsigned*)(ws + (size_t)NM * DIM * 2
                                    + (size_t)NQ * DIM * 2);           // 25 KB

    normalize_rows<<<NM, 256, 0, stream>>>(bank, bn, NM);
    normalize_rows<<<NQ, 256, 0, stream>>>(patch, qn, NQ);
    init_min<<<(NQ + 255) / 256, 256, 0, stream>>>(qmin, NQ);

    dim3 grid(QTILES, MCHUNKS);
    patchcore_nn<<<grid, 256, 0, stream>>>(qn, bn, qmin);

    topk_mean<<<NB, 256, 0, stream>>>(qmin, out);
}
